// MoE_multirules_38208029065733
// MI455X (gfx1250) — matrix operand order repaired and checked by reading, not compiled, not run on hardware
//
#include <hip/hip_runtime.h>

// ---------------------------------------------------------------------------
// MoE multirules for MI455X (gfx1250, wave32, WMMA bf16 16x16x32)
// B=16384, D=1024, C=1024
// ---------------------------------------------------------------------------

#define BATCH 16384
#define DDIM  1024
#define CDIM  1024

typedef __attribute__((ext_vector_type(16))) __bf16 v16bf;
typedef __attribute__((ext_vector_type(8)))  float  v8f;

union Frag {
    uint4 q[2];   // 32 bytes
    v16bf v;      // 16 x bf16
};

__device__ __forceinline__ unsigned short f2bf(float f) {
    unsigned int u = __float_as_uint(f);
    u += 0x7FFFu + ((u >> 16) & 1u);          // round-to-nearest-even
    return (unsigned short)(u >> 16);
}

// ---------------------------------------------------------------------------
// Kernel 1: transpose + convert weights to bf16:  WT[n][k] = bf16(W[k][n])
// grid (32, 32, 2), block 256
// ---------------------------------------------------------------------------
__global__ void __launch_bounds__(256) wt_kernel(const float* __restrict__ Wr,
                                                 const float* __restrict__ Wml,
                                                 unsigned short* __restrict__ WTr,
                                                 unsigned short* __restrict__ WTml) {
    const float* W = blockIdx.z ? Wml : Wr;
    unsigned short* WT = blockIdx.z ? WTml : WTr;
    __shared__ float tile[32][33];
    const int n0 = blockIdx.x * 32;
    const int k0 = blockIdx.y * 32;
    const int tx = threadIdx.x & 31;
    const int ty = threadIdx.x >> 5;          // 0..7
#pragma unroll
    for (int i = 0; i < 4; ++i)
        tile[ty + i * 8][tx] = W[(size_t)(k0 + ty + i * 8) * CDIM + n0 + tx];
    __syncthreads();
#pragma unroll
    for (int i = 0; i < 4; ++i)
        WT[(size_t)(n0 + ty + i * 8) * DDIM + (k0 + tx)] = f2bf(tile[tx][ty + i * 8]);
}

// ---------------------------------------------------------------------------
// Kernel 2: gate = softmax(x @ Wg + bg).  One wave per row (coalesced k).
// grid 2048, block 256 (8 waves)
// ---------------------------------------------------------------------------
__global__ void __launch_bounds__(256) gate_kernel(const float* __restrict__ x,
                                                   const float* __restrict__ Wg,
                                                   const float* __restrict__ bg,
                                                   float* __restrict__ gate_out) {
    const int wave = threadIdx.x >> 5;
    const int lane = threadIdx.x & 31;
    const int row  = blockIdx.x * 8 + wave;
    const float* xr = x + (size_t)row * DDIM;
    float a0 = 0.f, a1 = 0.f;
    for (int k = lane; k < DDIM; k += 32) {
        const float xv = xr[k];
        a0 = fmaf(xv, Wg[2 * k + 0], a0);
        a1 = fmaf(xv, Wg[2 * k + 1], a1);
    }
#pragma unroll
    for (int m = 16; m >= 1; m >>= 1) {
        a0 += __shfl_xor(a0, m, 32);
        a1 += __shfl_xor(a1, m, 32);
    }
    if (lane == 0) {
        a0 += bg[0];
        a1 += bg[1];
        const float mx = fmaxf(a0, a1);
        const float e0 = expf(a0 - mx);
        const float e1 = expf(a1 - mx);
        const float inv = 1.f / (e0 + e1);
        gate_out[2 * row + 0] = e0 * inv;
        gate_out[2 * row + 1] = e1 * inv;
    }
}

// ---------------------------------------------------------------------------
// Kernel 3: main fused GEMM kernel.
// One WG = 64 rows x all 1024 cols.  8 waves; each wave owns 4 row-blocks x
// 2 n-tiles per step (8 accumulators), 4 steps over N.
// Sweep 0: y_rule (relu(x@Wr+br)*smask) -> stored into y[] temporarily,
//          per-row sums -> mask_support (deterministic LDS reduction).
// Sweep 1: y_ml = x@Wml+bml, reload y_rule, apply gate mixing, write final y.
// ---------------------------------------------------------------------------
#define XSTR 1032            // ushorts per staged-x row (1024 + 8 pad) = 516 dwords

__global__ void __launch_bounds__(256) moe_main(const float* __restrict__ x,
                                                const float* __restrict__ bml,
                                                const float* __restrict__ br,
                                                const float* __restrict__ gate,
                                                const int*   __restrict__ smask,
                                                const unsigned short* __restrict__ WTr,
                                                const unsigned short* __restrict__ WTml,
                                                float* __restrict__ y,
                                                float* __restrict__ mask_out) {
    __shared__ __align__(16) unsigned short xs[64 * XSTR];   // 132 KB staged bf16 x
    __shared__ float partial[8 * 64];                        // per-wave row partial sums
    __shared__ float masks[64];
    __shared__ float g0l[64], g1l[64], smf[64];
    __shared__ float brl[CDIM], bmll[CDIM];

    const int tid    = threadIdx.x;
    const int wgrow0 = blockIdx.x * 64;

    // ---- stage x (fp32 -> bf16), gate scalars, biases ----
    for (int i = tid; i < 64 * DDIM; i += 256) {
        const int r = i >> 10, k = i & 1023;
        xs[r * XSTR + k] = f2bf(x[(size_t)(wgrow0 + r) * DDIM + k]);
    }
    if (tid < 64) {
        g0l[tid] = gate[(wgrow0 + tid) * 2 + 0];
        g1l[tid] = gate[(wgrow0 + tid) * 2 + 1];
        smf[tid] = (float)smask[wgrow0 + tid];
    }
    for (int i = tid; i < 512; i += 256) partial[i] = 0.f;
    for (int i = tid; i < CDIM; i += 256) { brl[i] = br[i]; bmll[i] = bml[i]; }
    __syncthreads();

    const int wave  = tid >> 5;
    const int lane  = tid & 31;
    const int col16 = lane & 15;
    const int hi    = lane >> 4;        // 0: lanes 0-15, 1: lanes 16-31
    const int aoffd = hi * 8;           // A-frag dword offset (K 0-15 vs 16-31)
    const int kselB = hi * 16;          // B-frag element offset (K 0-15 vs 16-31)
    const unsigned int* xsd = (const unsigned int*)xs;

    for (int sweep = 0; sweep < 2; ++sweep) {
        const unsigned short* WT = (sweep == 0) ? WTr : WTml;

        for (int it = 0; it < 4; ++it) {
            const int n0 = it * 256 + wave * 32;
            const unsigned short* wbase0 = WT + (size_t)(n0 + col16) * DDIM + kselB;
            const unsigned short* wbase1 = wbase0 + 16 * DDIM;

            v8f acc[4][2];
#pragma unroll
            for (int r = 0; r < 4; ++r) { acc[r][0] = (v8f)0.f; acc[r][1] = (v8f)0.f; }

            for (int k0 = 0; k0 < DDIM; k0 += 32) {
                Frag b0, b1;
                const uint4* pb0 = (const uint4*)(wbase0 + k0);
                b0.q[0] = pb0[0]; b0.q[1] = pb0[1];
                const uint4* pb1 = (const uint4*)(wbase1 + k0);
                b1.q[0] = pb1[0]; b1.q[1] = pb1[1];
#pragma unroll
                for (int r = 0; r < 4; ++r) {
                    Frag a;
                    const unsigned int* xr =
                        xsd + (unsigned)((r * 16 + col16) * (XSTR / 2) + (k0 >> 1) + aoffd);
                    a.q[0] = *(const uint4*)(xr);
                    a.q[1] = *(const uint4*)(xr + 4);
                    acc[r][0] = __builtin_amdgcn_wmma_f32_16x16x32_bf16(
                        false, a.v, false, b0.v, (short)0, acc[r][0], false, false);
                    acc[r][1] = __builtin_amdgcn_wmma_f32_16x16x32_bf16(
                        false, a.v, false, b1.v, (short)0, acc[r][1], false, false);
                }
            }

            // ---- epilogue for the 8 tiles of this step ----
#pragma unroll
            for (int r = 0; r < 4; ++r) {
#pragma unroll
                for (int t = 0; t < 2; ++t) {
                    const int col = n0 + t * 16 + col16;
                    const int lrb = r * 16 + hi * 8;      // local row base for VGPR j=0
                    v8f v = acc[r][t];
                    if (sweep == 0) {
                        const float bc = brl[col];
#pragma unroll
                        for (int j = 0; j < 8; ++j) {
                            const float val = fmaxf(v[j] + bc, 0.f) * smf[lrb + j];
                            v[j] = val;
                            y[(size_t)(wgrow0 + lrb + j) * CDIM + col] = val;
                        }
                        // reduce over the 16 columns (stays within each lane half)
#pragma unroll
                        for (int m = 1; m <= 8; m <<= 1) {
#pragma unroll
                            for (int j = 0; j < 8; ++j) v[j] += __shfl_xor(v[j], m, 32);
                        }
                        if (col16 == 0) {
#pragma unroll
                            for (int j = 0; j < 8; ++j)
                                partial[wave * 64 + lrb + j] += v[j];
                        }
                    } else {
                        const float bc = bmll[col];
#pragma unroll
                        for (int j = 0; j < 8; ++j) {
                            const int lrow = lrb + j;
                            const size_t gi = (size_t)(wgrow0 + lrow) * CDIM + col;
                            const float yml = v[j] + bc;
                            const float yr  = y[gi];       // y_rule parked here in sweep 0
                            float s0, s1;
                            if (masks[lrow] != 0.f) {
                                s0 = g0l[lrow];
                                s1 = ((wgrow0 + lrow) == 0) ? g1l[lrow]
                                                            : g0l[lrow] * g1l[lrow];
                            } else {
                                s0 = 1.f; s1 = 0.f;
                            }
                            y[gi] = s0 * yml + s1 * yr;
                        }
                    }
                }
            }
        }

        if (sweep == 0) {
            __threadfence();
            __syncthreads();
            if (tid < 64) {
                float s = 0.f;
#pragma unroll
                for (int w = 0; w < 8; ++w) s += partial[w * 64 + tid];
                masks[tid] = s;
                mask_out[wgrow0 + tid] = s;
            }
            __syncthreads();
        }
    }
}

// ---------------------------------------------------------------------------
// Launch: inputs are (x, Wg, bg, Wml, bml, Wr, br, support_mask).
// d_out = [ y (B*C) | gate (B*2) | mask_support (B) ]  (fp32, flat, in order)
// d_ws  = bf16 transposed weights: WTr then WTml (4 MB total).
// ---------------------------------------------------------------------------
extern "C" void kernel_launch(void* const* d_in, const int* in_sizes, int n_in,
                              void* d_out, int out_size, void* d_ws, size_t ws_size,
                              hipStream_t stream) {
    const float* x    = (const float*)d_in[0];
    const float* Wg   = (const float*)d_in[1];
    const float* bg   = (const float*)d_in[2];
    const float* Wml  = (const float*)d_in[3];
    const float* bml  = (const float*)d_in[4];
    const float* Wr   = (const float*)d_in[5];
    const float* br   = (const float*)d_in[6];
    const int*   sm   = (const int*)d_in[7];

    float* out     = (float*)d_out;
    float* y       = out;
    float* gateout = out + (size_t)BATCH * CDIM;
    float* maskout = gateout + (size_t)BATCH * 2;

    unsigned short* WTr  = (unsigned short*)d_ws;
    unsigned short* WTml = WTr + (size_t)DDIM * CDIM;

    wt_kernel<<<dim3(CDIM / 32, DDIM / 32, 2), 256, 0, stream>>>(Wr, Wml, WTr, WTml);
    gate_kernel<<<dim3(BATCH / 8), 256, 0, stream>>>(x, Wg, bg, gateout);
    moe_main<<<dim3(BATCH / 64), 256, 0, stream>>>(x, bml, br, gateout, sm,
                                                   WTr, WTml, y, maskout);
}
